// AsynchronousGNN_84421877170710
// MI455X (gfx1250) — matrix operand order repaired and checked by reading, not compiled, not run on hardware
//
#include <hip/hip_runtime.h>
#include <hip/hip_bf16.h>

// ---------------------------------------------------------------------------
// CDNA5 (gfx1250) GNN pipeline: bf16 WMMA GEMMs (v_wmma_f32_16x16x32_bf16),
// L2-resident gather/scatter (node table 25.6MB << 192MB L2), f32 atomics.
// ---------------------------------------------------------------------------

typedef __bf16 bf16_t;
typedef bf16_t v16bf __attribute__((ext_vector_type(16)));
typedef bf16_t v8bf  __attribute__((ext_vector_type(8)));
typedef float  v8f   __attribute__((ext_vector_type(8)));

#define N_NODES   50000
#define N_EDGES_  800000
#define HID       128
#define EDGE_HID  64
#define IN_NF     64
#define OUT_NF    64

__device__ __forceinline__ bf16_t f2bf(float x) {
    union { float f; unsigned u; } in; in.f = x;
    unsigned r = in.u + 0x7FFFu + ((in.u >> 16) & 1u);   // round-to-nearest-even
    union { unsigned short s; bf16_t b; } out;
    out.s = (unsigned short)(r >> 16);
    return out.b;
}

__device__ __forceinline__ float silu(float x) {
    return x / (1.0f + __expf(-x));
}

// A-operand fragment (16-bit A 16x32 layout): lane half selects K offset 0/8;
// elements 0..7 = K[k0+kh*8 .. +7], elements 8..15 = K[k0+16+kh*8 .. +7].
__device__ __forceinline__ v16bf load_a_frag(const bf16_t* row, int k0, int khalf) {
    const bf16_t* p = row + k0 + khalf * 8;
    v8bf lo = *(const v8bf*)(p);
    v8bf hi = *(const v8bf*)(p + 16);
    v16bf f;
#pragma unroll
    for (int i = 0; i < 8; ++i) { f[i] = lo[i]; f[i + 8] = hi[i]; }
    return f;
}

// B-operand fragment (16-bit B 32x16 layout): lane half selects K offset 0/8;
// elements 0..7 = K[k0+kh*8 .. +7], elements 8..15 = K[k0+16+kh*8 .. +7].
__device__ __forceinline__ v16bf load_b_frag(const bf16_t* colp, int k0, int khalf) {
    const bf16_t* p = colp + k0 + khalf * 8;
    v8bf lo = *(const v8bf*)(p);
    v8bf hi = *(const v8bf*)(p + 16);
    v16bf f;
#pragma unroll
    for (int i = 0; i < 8; ++i) { f[i] = lo[i]; f[i + 8] = hi[i]; }
    return f;
}

__device__ __forceinline__ v8f wmma_bf16(v16bf a, v16bf b, v8f c) {
    return __builtin_amdgcn_wmma_f32_16x16x32_bf16(false, a, false, b,
                                                   (short)0, c, false, false);
}

// ---------------------------------------------------------------------------
// Weight prep: fp32 [K][N] row-major -> bf16 transposed [N][K]
// ---------------------------------------------------------------------------
__global__ void convert_transpose_kernel(const float* __restrict__ src,
                                         bf16_t* __restrict__ dst, int K, int N) {
    long total = (long)K * N;
    for (long idx = (long)blockIdx.x * blockDim.x + threadIdx.x; idx < total;
         idx += (long)gridDim.x * blockDim.x) {
        int k = (int)(idx / N), n = (int)(idx % N);
        dst[(long)n * K + k] = f2bf(src[idx]);
    }
}

__global__ void zero_kernel(float4* __restrict__ p, long n4) {
    for (long i = (long)blockIdx.x * blockDim.x + threadIdx.x; i < n4;
         i += (long)gridDim.x * blockDim.x)
        p[i] = float4{0.f, 0.f, 0.f, 0.f};
}

// ---------------------------------------------------------------------------
// Dense: Y[N][NOUT] = X[N][KIN] @ W + b   (embeddings).  64 rows/block, 4 waves.
// ---------------------------------------------------------------------------
template <int KIN, int NOUT>
__global__ __launch_bounds__(128) void dense_kernel(
    const float* __restrict__ X, const bf16_t* __restrict__ WT,
    const float* __restrict__ bias, float* __restrict__ Y, int nRows) {
    __shared__ __align__(16) bf16_t Abuf[64 * KIN];
    const int t = threadIdx.x, lane = t & 31, wave = t >> 5;
    const int r = lane & 15, khalf = lane >> 4;
    const int nTiles = (nRows + 63) >> 6;
    constexpr int NT = NOUT / 16;
    for (int tile = blockIdx.x; tile < nTiles; tile += gridDim.x) {
        __syncthreads();
        for (int idx = t; idx < 64 * (KIN / 4); idx += 128) {
            int rw = idx / (KIN / 4), q = idx % (KIN / 4);
            int node = tile * 64 + rw;
            float4 v = {0.f, 0.f, 0.f, 0.f};
            if (node < nRows) v = *(const float4*)(X + (long)node * KIN + q * 4);
            bf16_t* d = &Abuf[rw * KIN + q * 4];
            d[0] = f2bf(v.x); d[1] = f2bf(v.y); d[2] = f2bf(v.z); d[3] = f2bf(v.w);
        }
        __syncthreads();
        v8f acc[NT] = {};
        const bf16_t* Arow = &Abuf[(wave * 16 + r) * KIN];
#pragma unroll
        for (int k0 = 0; k0 < KIN; k0 += 32) {
            v16bf a = load_a_frag(Arow, k0, khalf);
#pragma unroll
            for (int n = 0; n < NT; ++n) {
                v16bf b = load_b_frag(WT + (long)(n * 16 + r) * KIN, k0, khalf);
                acc[n] = wmma_bf16(a, b, acc[n]);
            }
        }
#pragma unroll
        for (int n = 0; n < NT; ++n) {
            int col = n * 16 + r;
            float bb = bias[col];
#pragma unroll
            for (int v = 0; v < 8; ++v) {
                int node = tile * 64 + wave * 16 + khalf * 8 + v;
                if (node < nRows) Y[(long)node * NOUT + col] = acc[n][v] + bb;
            }
        }
    }
}

// ---------------------------------------------------------------------------
// Edge model: gather -> silu(feat@ew1+b) -> silu(@ew2+b)*C -> atomic scatter
// 64 edges/chunk, 4 waves x 16 edges each.
// ---------------------------------------------------------------------------
__global__ __launch_bounds__(128) void edge_kernel(
    const float* __restrict__ h, const int* __restrict__ rowIdx,
    const int* __restrict__ colIdx,
    const bf16_t* __restrict__ ew1T, const float* __restrict__ eb1,
    const bf16_t* __restrict__ ew2T, const float* __restrict__ eb2,
    float* __restrict__ agg, float C, int nEdges) {
    __shared__ __align__(16) bf16_t feat[64 * 256];   // 32 KB
    __shared__ __align__(16) bf16_t m1[64 * 64];      //  8 KB
    __shared__ int eRow[64];
    __shared__ int eCol[64];
    const int t = threadIdx.x, lane = t & 31, wave = t >> 5;
    const int r = lane & 15, khalf = lane >> 4;
    for (long base = (long)blockIdx.x * 64; base < nEdges;
         base += (long)gridDim.x * 64) {
        __syncthreads();
        if (t < 64) {
            long e = base + t;
            eRow[t] = (e < nEdges) ? rowIdx[e] : 0;
            eCol[t] = (e < nEdges) ? colIdx[e] : 0;
        }
        __syncthreads();
        // gather: 64 edges x 64 float4 (h[row] | h[col]) -> bf16 LDS
        for (int idx = t; idx < 64 * 64; idx += 128) {
            int e = idx >> 6, q = idx & 63;
            int node = (q < 32) ? eRow[e] : eCol[e];
            float4 v = *(const float4*)(h + (long)node * HID + (q & 31) * 4);
            bf16_t* d = &feat[e * 256 + q * 4];
            d[0] = f2bf(v.x); d[1] = f2bf(v.y); d[2] = f2bf(v.z); d[3] = f2bf(v.w);
        }
        __syncthreads();
        // GEMM1: [64x256] @ [256x64]
        v8f acc[4] = {};
        const bf16_t* Arow = &feat[(wave * 16 + r) * 256];
#pragma unroll
        for (int k0 = 0; k0 < 256; k0 += 32) {
            v16bf a = load_a_frag(Arow, k0, khalf);
#pragma unroll
            for (int n = 0; n < 4; ++n) {
                v16bf b = load_b_frag(ew1T + (long)(n * 16 + r) * 256, k0, khalf);
                acc[n] = wmma_bf16(a, b, acc[n]);
            }
        }
#pragma unroll
        for (int n = 0; n < 4; ++n) {
            int col = n * 16 + r;
            float bb = eb1[col];
#pragma unroll
            for (int v = 0; v < 8; ++v)
                m1[(wave * 16 + khalf * 8 + v) * 64 + col] = f2bf(silu(acc[n][v] + bb));
        }
        __syncthreads();
        // GEMM2: [64x64] @ [64x64]
        v8f acc2[4] = {};
        const bf16_t* A2 = &m1[(wave * 16 + r) * 64];
#pragma unroll
        for (int k0 = 0; k0 < 64; k0 += 32) {
            v16bf a = load_a_frag(A2, k0, khalf);
#pragma unroll
            for (int n = 0; n < 4; ++n) {
                v16bf b = load_b_frag(ew2T + (long)(n * 16 + r) * 64, k0, khalf);
                acc2[n] = wmma_bf16(a, b, acc2[n]);
            }
        }
        // scatter-add (resolved in L2; agg fits in 192MB L2)
#pragma unroll
        for (int n = 0; n < 4; ++n) {
            int col = n * 16 + r;
            float bb = eb2[col];
#pragma unroll
            for (int v = 0; v < 8; ++v) {
                int eL = wave * 16 + khalf * 8 + v;
                if (base + eL < nEdges) {
                    float m = C * silu(acc2[n][v] + bb);
                    atomicAdd(&agg[(long)eRow[eL] * EDGE_HID + col], m);
                }
            }
        }
    }
}

// ---------------------------------------------------------------------------
// Node model: silu([h|agg]@nw1+b)@nw2+b, residual, masked in-place update.
// ---------------------------------------------------------------------------
__global__ __launch_bounds__(128) void node_kernel(
    float* __restrict__ h, const float* __restrict__ agg,
    const unsigned char* __restrict__ mask,
    const bf16_t* __restrict__ nw1T, const float* __restrict__ nb1,
    const bf16_t* __restrict__ nw2T, const float* __restrict__ nb2, int nNodes) {
    __shared__ __align__(16) bf16_t A1[64 * 192];    // 24 KB
    __shared__ __align__(16) bf16_t hid[64 * 128];   // 16 KB
    const int t = threadIdx.x, lane = t & 31, wave = t >> 5;
    const int r = lane & 15, khalf = lane >> 4;
    const int nTiles = (nNodes + 63) >> 6;
    for (int tile = blockIdx.x; tile < nTiles; tile += gridDim.x) {
        __syncthreads();
        // build [h | agg] : 64 rows x (128+64)
        for (int idx = t; idx < 64 * 48; idx += 128) {
            int rw = idx / 48, q = idx % 48;
            int node = tile * 64 + rw;
            float4 v = {0.f, 0.f, 0.f, 0.f};
            if (node < nNodes)
                v = (q < 32) ? *(const float4*)(h + (long)node * HID + q * 4)
                             : *(const float4*)(agg + (long)node * EDGE_HID + (q - 32) * 4);
            bf16_t* d = &A1[rw * 192 + q * 4];
            d[0] = f2bf(v.x); d[1] = f2bf(v.y); d[2] = f2bf(v.z); d[3] = f2bf(v.w);
        }
        __syncthreads();
        // GEMM1: [64x192] @ [192x128]
        v8f acc[8] = {};
        const bf16_t* Arow = &A1[(wave * 16 + r) * 192];
#pragma unroll
        for (int k0 = 0; k0 < 192; k0 += 32) {
            v16bf a = load_a_frag(Arow, k0, khalf);
#pragma unroll
            for (int n = 0; n < 8; ++n) {
                v16bf b = load_b_frag(nw1T + (long)(n * 16 + r) * 192, k0, khalf);
                acc[n] = wmma_bf16(a, b, acc[n]);
            }
        }
#pragma unroll
        for (int n = 0; n < 8; ++n) {
            int col = n * 16 + r;
            float bb = nb1[col];
#pragma unroll
            for (int v = 0; v < 8; ++v)
                hid[(wave * 16 + khalf * 8 + v) * 128 + col] = f2bf(silu(acc[n][v] + bb));
        }
        __syncthreads();
        // GEMM2: [64x128] @ [128x128]
        v8f acc2[8] = {};
        const bf16_t* A2 = &hid[(wave * 16 + r) * 128];
#pragma unroll
        for (int k0 = 0; k0 < 128; k0 += 32) {
            v16bf a = load_a_frag(A2, k0, khalf);
#pragma unroll
            for (int n = 0; n < 8; ++n) {
                v16bf b = load_b_frag(nw2T + (long)(n * 16 + r) * 128, k0, khalf);
                acc2[n] = wmma_bf16(a, b, acc2[n]);
            }
        }
        // residual + masked update (in place; one lane owns each element)
#pragma unroll
        for (int n = 0; n < 8; ++n) {
            int col = n * 16 + r;
            float bb = nb2[col];
#pragma unroll
            for (int v = 0; v < 8; ++v) {
                int node = tile * 64 + wave * 16 + khalf * 8 + v;
                if (node < nNodes && mask[node]) {
                    float* p = &h[(long)node * HID + col];
                    *p = *p + acc2[n][v] + bb;
                }
            }
        }
    }
}

// ---------------------------------------------------------------------------
extern "C" void kernel_launch(void* const* d_in, const int* in_sizes, int n_in,
                              void* d_out, int out_size, void* d_ws, size_t ws_size,
                              hipStream_t stream) {
    const float* h_in      = (const float*)d_in[0];
    const int*   edges_a   = (const int*)d_in[1];
    const int*   edges_b   = (const int*)d_in[2];
    const unsigned char* mask_a = (const unsigned char*)d_in[3];
    const unsigned char* mask_b = (const unsigned char*)d_in[4];
    const float* emb_in_w  = (const float*)d_in[5];
    const float* emb_in_b  = (const float*)d_in[6];
    const float* emb_out_w = (const float*)d_in[7];
    const float* emb_out_b = (const float*)d_in[8];
    const float* edge_w1   = (const float*)d_in[9];
    const float* edge_b1   = (const float*)d_in[10];
    const float* edge_w2   = (const float*)d_in[11];
    const float* edge_b2   = (const float*)d_in[12];
    const float* node_w1   = (const float*)d_in[13];
    const float* node_b1   = (const float*)d_in[14];
    const float* node_w2   = (const float*)d_in[15];
    const float* node_b2   = (const float*)d_in[16];

    // workspace layout (all offsets 16B aligned)
    char* ws = (char*)d_ws;
    float*  hbuf = (float*)ws;                                   // 50000*128 f32
    float*  agg  = (float*)(ws + (size_t)N_NODES * HID * 4);     // 50000*64  f32
    bf16_t* wbf  = (bf16_t*)(ws + (size_t)N_NODES * HID * 4
                                + (size_t)N_NODES * EDGE_HID * 4);
    bf16_t* embInWT  = wbf;                 // [128][64]
    bf16_t* embOutWT = wbf + 64 * 128;      // [64][128]
    bf16_t* lwBase   = wbf + 64 * 128 + 128 * 64;
    const long LW_STRIDE = 256 * 64 + 64 * 64 + 192 * 128 + 128 * 128; // 61440

    auto cvgrid = [](long n) { return (int)((n + 255) / 256); };

    // ---- weight prep: transpose + bf16 convert ----
    convert_transpose_kernel<<<cvgrid(64 * 128), 256, 0, stream>>>(emb_in_w, embInWT, IN_NF, HID);
    convert_transpose_kernel<<<cvgrid(128 * 64), 256, 0, stream>>>(emb_out_w, embOutWT, HID, OUT_NF);
    for (int i = 0; i < 4; ++i) {
        bf16_t* lw = lwBase + (long)i * LW_STRIDE;
        bf16_t* ew1T = lw;
        bf16_t* ew2T = lw + 256 * 64;
        bf16_t* nw1T = lw + 256 * 64 + 64 * 64;
        bf16_t* nw2T = lw + 256 * 64 + 64 * 64 + 192 * 128;
        convert_transpose_kernel<<<cvgrid(256 * 64), 256, 0, stream>>>(edge_w1 + (long)i * 256 * 64, ew1T, 2 * HID, EDGE_HID);
        convert_transpose_kernel<<<cvgrid(64 * 64), 256, 0, stream>>>(edge_w2 + (long)i * 64 * 64, ew2T, EDGE_HID, EDGE_HID);
        convert_transpose_kernel<<<cvgrid(192 * 128), 256, 0, stream>>>(node_w1 + (long)i * 192 * 128, nw1T, HID + EDGE_HID, HID);
        convert_transpose_kernel<<<cvgrid(128 * 128), 256, 0, stream>>>(node_w2 + (long)i * 128 * 128, nw2T, HID, HID);
    }

    const int nTiles = (N_NODES + 63) / 64;   // 782

    // ---- input embedding ----
    dense_kernel<IN_NF, HID><<<nTiles, 128, 0, stream>>>(h_in, embInWT, emb_in_b, hbuf, N_NODES);

    // ---- 4 GCL layers ----
    for (int i = 0; i < 4; ++i) {
        bf16_t* lw = lwBase + (long)i * LW_STRIDE;
        bf16_t* ew1T = lw;
        bf16_t* ew2T = lw + 256 * 64;
        bf16_t* nw1T = lw + 256 * 64 + 64 * 64;
        bf16_t* nw2T = lw + 256 * 64 + 64 * 64 + 192 * 128;
        const int* edges = (i % 2 == 0) ? edges_a : edges_b;
        const unsigned char* mask = (i % 2 == 0) ? mask_a : mask_b;
        float C = (i % 2 == 0) ? 1.0f : (2.0f / 64.0f);

        zero_kernel<<<3125, 256, 0, stream>>>((float4*)agg, (long)N_NODES * EDGE_HID / 4);
        edge_kernel<<<1250, 128, 0, stream>>>(hbuf, edges, edges + N_EDGES_,
                                              ew1T, edge_b1 + (long)i * 64,
                                              ew2T, edge_b2 + (long)i * 64,
                                              agg, C, N_EDGES_);
        node_kernel<<<nTiles, 128, 0, stream>>>(hbuf, agg, mask,
                                                nw1T, node_b1 + (long)i * 128,
                                                nw2T, node_b2 + (long)i * 128, N_NODES);
    }

    // ---- output embedding ----
    dense_kernel<HID, OUT_NF><<<nTiles, 128, 0, stream>>>(hbuf, embOutWT, emb_out_b,
                                                          (float*)d_out, N_NODES);
}
